// BatchedNeRFMLP_47141561041600
// MI455X (gfx1250) — matrix operand order repaired and checked by reading, not compiled, not run on hardware
//
#include <hip/hip_runtime.h>

typedef __attribute__((ext_vector_type(16))) _Float16 v16h;
typedef __attribute__((ext_vector_type(8)))  _Float16 v8h;
typedef __attribute__((ext_vector_type(8)))  float    v8f;

namespace {
constexpr int kB     = 64;
constexpr int kN     = 16384;
constexpr int kHid   = 128;
constexpr int kPosIn = 63;
constexpr int kKpos  = 64;    // 63 padded to 64
constexpr int kDirIn = 27;
constexpr int kKhead = 160;   // 128 h + 27 dir + 1 bias-feature + 4 pad
constexpr int kTile  = 128;   // points per tile
constexpr int kIters = 4;     // tiles per block -> 512 points/block
constexpr int kTotal = 8789;
// param offsets within one batch's parameter vector
constexpr int OFF_PW = 0;     // 128*63
constexpr int OFF_PB = 8064;  // 128
constexpr int OFF_SW = 8192;  // 128
constexpr int OFF_SB = 8320;  // 1
constexpr int OFF_CW = 8321;  // 3*155
constexpr int OFF_CB = 8786;  // 3
}

__global__ __launch_bounds__(256)
void nerf_mlp_wmma_f16(const float* __restrict__ pos,
                       const float* __restrict__ dirs,
                       const float* __restrict__ prm,
                       float* __restrict__ out)
{
  // Static LDS: 16384 (Wt -> A_pos union) + 40960 (Hbuf) + 5120 (head W) + 512 = 62976 B
  __shared__ __align__(32) _Float16 sA [kTile * kKpos];   // staged Wt[hid][k], then A_pos[p][64]
  __shared__ __align__(32) _Float16 sH [kTile * kKhead];  // Hbuf [p][160]
  __shared__ __align__(32) _Float16 sWh[16 * kKhead];     // head weights [col][k]
  __shared__ float sPb[kHid];

  const int b       = blockIdx.y;
  const int t       = threadIdx.x;
  const int lane    = t & 31;          // wave32
  const int wave    = t >> 5;          // 8 waves
  const int m       = lane & 15;       // row (A) / col (B,D) index within 16x16 tile
  const int kh      = lane >> 4;       // K half-select per WMMA operand layout
  const int rowbase = wave * 16;       // wave's hidden-col tile (GEMM) / point rows (head)
  const int blockBase = b * kN + blockIdx.x * (kTile * kIters);

  const float* __restrict__ P = prm + (size_t)b * kTotal;
  float* __restrict__ outSigma = out;
  float* __restrict__ outRgb   = out + (size_t)kB * kN;

  // ---------------- Phase 0: stage per-batch weights into LDS (f16), once ----------
  for (int idx = t; idx < kHid * kKpos; idx += 256) {
    const int h = idx >> 6, k = idx & 63;
    const float v = (k < kPosIn) ? P[OFF_PW + h * kPosIn + k] : 0.0f;
    sA[idx] = (_Float16)v;                       // Wt[hid][k]
  }
  if (t < kHid) sPb[t] = P[OFF_PB + t];
  for (int idx = t; idx < 16 * kKhead; idx += 256) {
    const int col = idx / kKhead;
    const int k   = idx - col * kKhead;
    float v = 0.0f;
    if (col == 0) {                              // sigma: sw over h, bias at k=155
      if (k < kHid)       v = P[OFF_SW + k];
      else if (k == 155)  v = P[OFF_SB];
    } else if (col <= 3) {                       // rgb: cw over [h|dir], bias at k=155
      const int c = col - 1;
      if (k < 155)        v = P[OFF_CW + c * 155 + k];
      else if (k == 155)  v = P[OFF_CB + c];
    }
    sWh[idx] = (_Float16)v;
  }
  __syncthreads();

  // ------- Hoist B operands into registers (held for the whole kernel) -------------
  // Main GEMM: wave owns hidden cols [rowbase, rowbase+16). B layout: lane col n=m
  // holds 16 contiguous K at kbase + 16*kh.
  const int hid = rowbase + m;
  const v16h B0  = *(const v16h*)(sA + hid * kKpos +      16 * kh);
  const v16h B1  = *(const v16h*)(sA + hid * kKpos + 32 + 16 * kh);
  const float pbv = sPb[hid];
  v16h Bh[5];                                    // head B, K=160 in 5 K-steps
  #pragma unroll
  for (int s = 0; s < 5; ++s)
    Bh[s] = *(const v16h*)(sWh + m * kKhead + s * 32 + 16 * kh);

  if (t >= kTile) {                              // constant tail cols of Hbuf, once
    _Float16* hr = sH + (t - kTile) * kKhead;
    hr[155] = (_Float16)1.0f;                    // bias feature
    hr[156] = (_Float16)0.0f; hr[157] = (_Float16)0.0f;
    hr[158] = (_Float16)0.0f; hr[159] = (_Float16)0.0f;
  }
  __syncthreads();                               // Wt in sA now dead -> A_pos region

  // =================== Tile loop: 4 x 128 points per block ========================
  #pragma unroll 1
  for (int it = 0; it < kIters; ++it) {
    const int pbase = blockBase + it * kTile;    // flattened b*N + n for this tile

    // -------- encodings: waves 0-3 pos_enc -> sA; waves 4-7 dir_enc -> Hbuf -------
    if (t < kTile) {
      const float* xp = pos + (size_t)(pbase + t) * 3;
      const float x0 = xp[0], x1 = xp[1], x2 = xp[2];
      _Float16* row = sA + t * kKpos;
      row[0] = (_Float16)x0; row[1] = (_Float16)x1; row[2] = (_Float16)x2;
      float f = 3.14159265358979323846f;
      #pragma unroll
      for (int fr = 0; fr < 10; ++fr) {
        const int base = 3 + fr * 6;
        row[base + 0] = (_Float16)__sinf(x0 * f);
        row[base + 1] = (_Float16)__sinf(x1 * f);
        row[base + 2] = (_Float16)__sinf(x2 * f);
        row[base + 3] = (_Float16)__cosf(x0 * f);
        row[base + 4] = (_Float16)__cosf(x1 * f);
        row[base + 5] = (_Float16)__cosf(x2 * f);
        f *= 2.0f;
      }
      row[63] = (_Float16)0.0f;                  // K pad
    } else {
      const int p = t - kTile;
      const float* xd = dirs + (size_t)(pbase + p) * 3;
      const float x0 = xd[0], x1 = xd[1], x2 = xd[2];
      _Float16* hr = sH + p * kKhead + kHid;     // cols 128..154
      hr[0] = (_Float16)x0; hr[1] = (_Float16)x1; hr[2] = (_Float16)x2;
      float f = 3.14159265358979323846f;
      #pragma unroll
      for (int fr = 0; fr < 4; ++fr) {
        const int base = 3 + fr * 6;
        hr[base + 0] = (_Float16)__sinf(x0 * f);
        hr[base + 1] = (_Float16)__sinf(x1 * f);
        hr[base + 2] = (_Float16)__sinf(x2 * f);
        hr[base + 3] = (_Float16)__cosf(x0 * f);
        hr[base + 4] = (_Float16)__cosf(x1 * f);
        hr[base + 5] = (_Float16)__cosf(x2 * f);
        f *= 2.0f;
      }
    }
    __syncthreads();

    // -------- main GEMM: all 128 points x this wave's 16 hidden cols --------------
    // A as loaded here (contiguous K, same as B): lanes 0-15 hold K {0..15}(+kbase),
    // lanes 16-31 hold {16..31}(+kbase). Row stride = 64 halves.
    // Software-pipelined 1 stage deep: the ReLU/convert/store VALU of tile rt-1 is
    // emitted between the WMMA pairs of tile rt so it can co-execute in the slots
    // the F16 WMMA->VALU hazard would otherwise fill with v_nops.
    v8f accPrev;
    int  mmPrevBase = 0;
    #pragma unroll
    for (int rt = 0; rt < 8; ++rt) {
      const _Float16* arow = sA + (rt * 16 + m) * kKpos;
      const v8h a00 = *(const v8h*)(arow +      16 * kh);
      const v8h a01 = *(const v8h*)(arow +  8 + 16 * kh);
      const v8h a10 = *(const v8h*)(arow + 32 + 16 * kh);
      const v8h a11 = *(const v8h*)(arow + 40 + 16 * kh);
      v16h A0, A1;
      #pragma unroll
      for (int i = 0; i < 8; ++i) {
        A0[i] = a00[i]; A0[8 + i] = a01[i];
        A1[i] = a10[i]; A1[8 + i] = a11[i];
      }
      v8f acc = {pbv, pbv, pbv, pbv, pbv, pbv, pbv, pbv};  // bias pre-load (D col = m)
      acc = __builtin_amdgcn_wmma_f32_16x16x32_f16(false, A0, false, B0, (short)0, acc, false, false);
      acc = __builtin_amdgcn_wmma_f32_16x16x32_f16(false, A1, false, B1, (short)0, acc, false, false);
      if (rt > 0) {                               // drain previous tile (independent VALU)
        #pragma unroll
        for (int r = 0; r < 8; ++r) {
          const float x = accPrev[r];
          const float v = x > 0.0f ? x : 0.0f;   // single v_max_num_f32
          sH[(mmPrevBase + r + 8 * kh) * kKhead + rowbase + m] = (_Float16)v;
        }
      }
      accPrev    = acc;
      mmPrevBase = rt * 16;
    }
    #pragma unroll
    for (int r = 0; r < 8; ++r) {                // drain last tile
      const float x = accPrev[r];
      const float v = x > 0.0f ? x : 0.0f;
      sH[(mmPrevBase + r + 8 * kh) * kKhead + rowbase + m] = (_Float16)v;
    }
    __syncthreads();

    // -------- fused head GEMM (sigma + rgb), K = 160, rows = wave's 16 points -----
    v8f acc2 = {0.f, 0.f, 0.f, 0.f, 0.f, 0.f, 0.f, 0.f};  // biases via k=155 feature
    const _Float16* hrow = sH + (rowbase + m) * kKhead;
    #pragma unroll
    for (int s = 0; s < 5; ++s) {
      const int kb = s * 32;
      const v8h h0 = *(const v8h*)(hrow + kb +     16 * kh);
      const v8h h1 = *(const v8h*)(hrow + kb + 8 + 16 * kh);
      v16h Ah;
      #pragma unroll
      for (int i = 0; i < 8; ++i) { Ah[i] = h0[i]; Ah[8 + i] = h1[i]; }
      acc2 = __builtin_amdgcn_wmma_f32_16x16x32_f16(false, Ah, false, Bh[s], (short)0, acc2, false, false);
    }

    // D cols: 0 = sigma (raw), 1-3 = rgb (sigmoid). Divergence only after all WMMAs.
    if (m < 4) {
      #pragma unroll
      for (int r = 0; r < 8; ++r) {
        const int mm = r + 8 * kh;
        const int np = pbase + rowbase + mm;     // flattened b*N + n
        const float v = acc2[r];
        if (m == 0) {
          outSigma[np] = v;
        } else {
          // sigmoid via v_exp_f32 + v_rcp_f32 (approximate activation path)
          outRgb[(size_t)np * 3 + (m - 1)] = __builtin_amdgcn_rcpf(1.0f + __expf(-v));
        }
      }
    }
    __syncthreads();   // Hbuf/A region reads done before next tile's writes
  }
}

extern "C" void kernel_launch(void* const* d_in, const int* in_sizes, int n_in,
                              void* d_out, int out_size, void* d_ws, size_t ws_size,
                              hipStream_t stream) {
  (void)in_sizes; (void)n_in; (void)out_size; (void)d_ws; (void)ws_size;
  const float* pos  = (const float*)d_in[0];
  const float* dirs = (const float*)d_in[1];
  const float* prm  = (const float*)d_in[2];
  float* out = (float*)d_out;
  dim3 grid(kN / (kTile * kIters), kB);   // 32 x 64 blocks, 512 points each
  dim3 block(256);                        // 8 wave32 waves
  nerf_mlp_wmma_f16<<<grid, block, 0, stream>>>(pos, dirs, prm, out);
}
